// MultiHeadDifferentialAttention_82660940579039
// MI455X (gfx1250) — matrix operand order repaired and checked by reading, not compiled, not run on hardware
//
#include <hip/hip_runtime.h>
#include <math.h>

// ---------------------------------------------------------------------------
// MultiHeadDifferentialAttention for MI455X (gfx1250), wave32 + WMMA bf16.
// B=2, S=2048, HID=2048, NH=16, HD=128, QKD=64.
// GEMMs: bf16 operands, async global->LDS double-buffered panels (ASYNCcnt),
// ds_load_b128 fragment reads, v_wmma_f32_16x16x32_bf16 inner loop.
// ---------------------------------------------------------------------------

#define B_    2
#define S_    2048
#define HID_  2048
#define NH_   16
#define HD_   128
#define QKD_  64
#define MROWS (B_ * S_)            // 4096
#define LAMBDA_INIT_F   0.35550906759096926f
#define ONE_MINUS_LI_F  0.64449093240903074f

typedef __bf16        v16bf __attribute__((ext_vector_type(16)));
typedef float         v8f   __attribute__((ext_vector_type(8)));
typedef float         f4    __attribute__((ext_vector_type(4)));
typedef unsigned int  u4    __attribute__((ext_vector_type(4)));
typedef int           i4    __attribute__((ext_vector_type(4)));

union B8 { u4 u; __bf16 h[8]; };

#ifndef __has_builtin
#define __has_builtin(x) 0
#endif

// ---- CDNA5 async global->LDS copy (tracked with ASYNCcnt) -----------------
#if __has_builtin(__builtin_amdgcn_global_load_async_to_lds_b128)
typedef __attribute__((address_space(1))) i4 gas_i4;
typedef __attribute__((address_space(3))) i4 las_i4;
__device__ __forceinline__ void async_ld_b128(const __bf16* g, __bf16* l) {
    __builtin_amdgcn_global_load_async_to_lds_b128((gas_i4*)g, (las_i4*)l, 0, 0);
}
#else
__device__ __forceinline__ void async_ld_b128(const __bf16* g, __bf16* l) {
    asm volatile("global_load_async_to_lds_b128 %0, %1, off"
                 :: "v"((unsigned)(unsigned long long)l),
                    "v"((unsigned long long)g)
                 : "memory");
}
#endif

#if __has_builtin(__builtin_amdgcn_s_wait_asynccnt)
#define WAIT_ASYNC(n) __builtin_amdgcn_s_wait_asynccnt(n)
#else
#define WAIT_ASYNC(n) asm volatile("s_wait_asynccnt " #n ::: "memory")
#endif

__device__ inline __bf16 tobf(float f) {
    unsigned u = __builtin_bit_cast(unsigned, f);
    unsigned r = (u + 0x7FFFu + ((u >> 16) & 1u)) >> 16;   // RNE
    unsigned short us = (unsigned short)r;
    return __builtin_bit_cast(__bf16, us);
}
__device__ inline float frombf(__bf16 h) {
    unsigned short us = __builtin_bit_cast(unsigned short, h);
    unsigned u = ((unsigned)us) << 16;
    return __builtin_bit_cast(float, u);
}

__device__ inline v8f wmma_bf16(v16bf a, v16bf b, v8f c) {
    return __builtin_amdgcn_wmma_f32_16x16x32_bf16(
        false, a, false, b, (short)0, c, false, false);
}

// A fragment (16x32 bf16): lane = M row, lane-half selects K sub-block.
// element e -> K = kc + 16*hi + e
__device__ inline v16bf ld_a_bf16(const __bf16* row, int kc, int hi) {
    B8 t0, t1;
    t0.u = *reinterpret_cast<const u4*>(row + kc + 16 * hi);
    t1.u = *reinterpret_cast<const u4*>(row + kc + 16 * hi + 8);
    v16bf a;
#pragma unroll
    for (int e = 0; e < 8; ++e) { a[e] = t0.h[e]; a[8 + e] = t1.h[e]; }
    return a;
}

// B fragment (32x16 bf16): lane = N col; element e -> K = 16*hi + e.
// p points at the 16 contiguous bf16 for this lane.
__device__ inline v16bf ld_b16(const __bf16* p) {
    B8 t0, t1;
    t0.u = *reinterpret_cast<const u4*>(p);
    t1.u = *reinterpret_cast<const u4*>(p + 8);
    v16bf b;
#pragma unroll
    for (int e = 0; e < 8; ++e) { b[e] = t0.h[e]; b[8 + e] = t1.h[e]; }
    return b;
}

// ---------------------------------------------------------------------------
// Streaming fp32 -> bf16 conversion, 8 elements/thread (b128 in, b128 out).
// ---------------------------------------------------------------------------
__global__ __launch_bounds__(256) void cvt_kernel(const float* __restrict__ in,
                                                  __bf16* __restrict__ out) {
    size_t i = ((size_t)blockIdx.x * 256 + threadIdx.x) * 8;
    f4 r0 = *reinterpret_cast<const f4*>(in + i);
    f4 r1 = *reinterpret_cast<const f4*>(in + i + 4);
    B8 t;
#pragma unroll
    for (int e = 0; e < 4; ++e) { t.h[e] = tobf(r0[e]); t.h[4 + e] = tobf(r1[e]); }
    *reinterpret_cast<u4*>(out + i) = t.u;
}

// ---------------------------------------------------------------------------
// GEMM: Y[M,2048] = X[M,2048] * W[2048,2048]^T  (torch Linear), bf16 operands.
// WG = 128 threads (4 waves). Block tile 64x128; wave tile 32x64 (2x4 WMMA).
// Per K-step (32): async-stage next A(64x32)/B(128x32) panels into LDS
// (double-buffered, ASYNCcnt), compute current panel from LDS.
// ---------------------------------------------------------------------------
__device__ __forceinline__ void stage_panel(const __bf16* __restrict__ X,
                                            const __bf16* __restrict__ W,
                                            int m0, int n0, int k0, int t,
                                            __bf16* Abuf, __bf16* Bbuf) {
    // A panel: 64 rows x 32 cols = 256 b128 chunks, 2 per thread
#pragma unroll
    for (int i = 0; i < 2; ++i) {
        int c = t + i * 128;
        int row = c >> 2, off = (c & 3) * 8;
        async_ld_b128(X + (size_t)(m0 + row) * HID_ + k0 + off,
                      Abuf + row * 32 + off);
    }
    // B panel: 128 rows x 32 cols = 512 b128 chunks, 4 per thread
#pragma unroll
    for (int i = 0; i < 4; ++i) {
        int c = t + i * 128;
        int row = c >> 2, off = (c & 3) * 8;
        async_ld_b128(W + (size_t)(n0 + row) * HID_ + k0 + off,
                      Bbuf + row * 32 + off);
    }
}

template <bool F32OUT>
__global__ __launch_bounds__(128) void gemm_bf16(const __bf16* __restrict__ X,
                                                 const __bf16* __restrict__ W,
                                                 void* __restrict__ Yv) {
    __shared__ __align__(16) __bf16 sA[2][64 * 32];    // 2 x 4 KiB
    __shared__ __align__(16) __bf16 sB[2][128 * 32];   // 2 x 8 KiB

    const int t    = threadIdx.x;
    const int lane = t & 31;
    const int w    = t >> 5;                 // 4 waves, 2x2 arrangement
    const int ln   = lane & 15;
    const int hi   = lane >> 4;
    const int NB   = HID_ / 128;             // 16 column blocks
    const int bm   = blockIdx.x / NB;
    const int bn   = blockIdx.x % NB;
    const int m0   = bm * 64;                // block row base
    const int n0   = bn * 128;               // block col base
    const int moff = (w >> 1) * 32;          // wave offset in A panel
    const int noff = (w & 1) * 64;           // wave offset in B panel

    v8f c[2][4];
#pragma unroll
    for (int i = 0; i < 2; ++i)
#pragma unroll
        for (int j = 0; j < 4; ++j)
#pragma unroll
            for (int e = 0; e < 8; ++e) c[i][j][e] = 0.0f;

    stage_panel(X, W, m0, n0, 0, t, sA[0], sB[0]);

    for (int kk = 0; kk < HID_ / 32; ++kk) {
        const int cur = kk & 1;
        if (kk + 1 < HID_ / 32) {
            stage_panel(X, W, m0, n0, (kk + 1) * 32, t, sA[1 - cur], sB[1 - cur]);
            WAIT_ASYNC(6);   // previous panel's 6 async ops complete
        } else {
            WAIT_ASYNC(0);
        }
        __syncthreads();     // all waves' staging of current panel visible

        v16bf a[2], bb[4];
#pragma unroll
        for (int mt = 0; mt < 2; ++mt)
            a[mt] = ld_a_bf16(&sA[cur][(moff + mt * 16 + ln) * 32], 0, hi);
#pragma unroll
        for (int nt = 0; nt < 4; ++nt)
            bb[nt] = ld_b16(&sB[cur][(noff + nt * 16 + ln) * 32 + 16 * hi]);
#pragma unroll
        for (int mt = 0; mt < 2; ++mt)
#pragma unroll
            for (int nt = 0; nt < 4; ++nt)
                c[mt][nt] = wmma_bf16(a[mt], bb[nt], c[mt][nt]);

        __syncthreads();     // reads done before buffer is overwritten
    }

#pragma unroll
    for (int mt = 0; mt < 2; ++mt)
#pragma unroll
        for (int nt = 0; nt < 4; ++nt)
#pragma unroll
            for (int v = 0; v < 8; ++v) {
                size_t row = (size_t)(m0 + moff + mt * 16 + v + 8 * hi);
                size_t col = (size_t)(n0 + noff + nt * 16 + ln);
                if (F32OUT)
                    ((float*)Yv)[row * HID_ + col] = c[mt][nt][v];
                else
                    ((__bf16*)Yv)[row * HID_ + col] = tobf(c[mt][nt][v]);
            }
}

// ---------------------------------------------------------------------------
// RoPE + head split: proj[B,S,HID] -> Qh/Kh [B,2,NH,S,QKD] (bf16).
// ---------------------------------------------------------------------------
__global__ __launch_bounds__(256) void rope_kernel(const __bf16* __restrict__ proj,
                                                   const float* __restrict__ cos_e,
                                                   const float* __restrict__ sin_e,
                                                   __bf16* __restrict__ outp) {
    size_t idx = (size_t)blockIdx.x * 256 + threadIdx.x;   // B*2*NH*S*32 pairs
    int p = (int)(idx & 31);
    size_t r = idx >> 5;
    int s = (int)(r & (S_ - 1));
    size_t r2 = r >> 11;
    int h = (int)(r2 & (NH_ - 1));
    size_t r3 = r2 >> 4;
    int g = (int)(r3 & 1);
    int b = (int)(r3 >> 1);

    const __bf16* in = proj + ((size_t)(b * S_ + s)) * HID_ + h * HD_ + g * QKD_ + 2 * p;
    float x0 = frombf(in[0]);
    float x1 = frombf(in[1]);
    float cc = cos_e[s * HD_ + 2 * p];
    float ss = sin_e[s * HD_ + 2 * p];
    __bf16* out = outp + ((((size_t)(b * 2 + g) * NH_ + h) * S_ + s) * QKD_) + 2 * p;
    out[0] = tobf(x0 * cc - x1 * ss);
    out[1] = tobf(x1 * cc + x0 * ss);
}

// Vt[b][h][d][s] = proj[b][s][h*128+d]  (bf16 transpose for PV B-operand)
__global__ __launch_bounds__(256) void vtrans_kernel(const __bf16* __restrict__ proj,
                                                     __bf16* __restrict__ Vt) {
    size_t idx = (size_t)blockIdx.x * 256 + threadIdx.x;   // B*NH*HD*S
    int s = (int)(idx & (S_ - 1));
    int d = (int)((idx >> 11) & (HD_ - 1));
    int h = (int)((idx >> 18) & (NH_ - 1));
    int b = (int)(idx >> 22);
    Vt[idx] = proj[((size_t)(b * S_ + s)) * HID_ + h * HD_ + d];
}

// lam = exp(<lq1,lk1>) - exp(<lq2,lk2>) + LAMBDA_INIT
__global__ __launch_bounds__(128) void lambda_kernel(const float* lq1, const float* lk1,
                                                     const float* lq2, const float* lk2,
                                                     float* lam) {
    __shared__ float s1[128], s2[128];
    int t = threadIdx.x;
    s1[t] = lq1[t] * lk1[t];
    s2[t] = lq2[t] * lk2[t];
    __syncthreads();
    for (int off = 64; off; off >>= 1) {
        if (t < off) { s1[t] += s1[t + off]; s2[t] += s2[t + off]; }
        __syncthreads();
    }
    if (t == 0) *lam = expf(s1[0]) - expf(s2[0]) + LAMBDA_INIT_F;
}

// ---------------------------------------------------------------------------
// Flash differential attention. 2 waves/WG: wave g handles score group g for
// one 16-query tile of one (b,h); combine + RMSNorm in the epilogue.
// Output stored bf16 so the Wo GEMM runs the pure-bf16 inner loop.
// ---------------------------------------------------------------------------
__global__ __launch_bounds__(64) void diffattn_kernel(const __bf16* __restrict__ Qh,
                                                      const __bf16* __restrict__ Kh,
                                                      const __bf16* __restrict__ Vt,
                                                      const float* __restrict__ lam_p,
                                                      const float* __restrict__ gnorm_w,
                                                      __bf16* __restrict__ O) {
    __shared__ __align__(16) __bf16 sP[2][16 * 32];   // P tile per group
    __shared__ float sO[2][16 * HD_];                 // per-group AV result
    __shared__ float sRed[64];

    const int lane = threadIdx.x & 31;
    const int g    = threadIdx.x >> 5;
    const int ln   = lane & 15;
    const int hi   = lane >> 4;
    const int blk  = blockIdx.x;                 // B*NH*(S/16) = 4096
    const int qi   = blk & 127;
    const int h    = (blk >> 7) & 15;
    const int b    = blk >> 11;
    const int m0   = qi * 16;

    const __bf16* Qbase = Qh + (((size_t)(b * 2 + g) * NH_ + h) * S_) * QKD_;
    const __bf16* Kbase = Kh + (((size_t)(b * 2 + g) * NH_ + h) * S_) * QKD_;
    const __bf16* Vbase = Vt + (((size_t)b * NH_ + h) * HD_) * S_;

    // Q fragments for K-dim chunks [0,32) and [32,64)
    const __bf16* qrow = Qbase + (size_t)(m0 + ln) * QKD_;
    v16bf qa0 = ld_a_bf16(qrow, 0, hi);
    v16bf qa1 = ld_a_bf16(qrow, 32, hi);

    v8f acc[8];
    float rmax[8], rsum[8];
#pragma unroll
    for (int f = 0; f < 8; ++f) {
#pragma unroll
        for (int e = 0; e < 8; ++e) acc[f][e] = 0.0f;
    }
#pragma unroll
    for (int v = 0; v < 8; ++v) { rmax[v] = -1e30f; rsum[v] = 0.0f; }

    const int ktend = (m0 + 15) >> 5;
    for (int kt = 0; kt <= ktend; ++kt) {
        const int kbase = kt * 32;

        // ---- scores: S[16,32] = Q[16,64] * K^T[64,32], two 16-col frags ----
        const __bf16* krow0 = Kbase + (size_t)(kbase + ln) * QKD_;
        const __bf16* krow1 = Kbase + (size_t)(kbase + 16 + ln) * QKD_;
        v8f s0, s1;
#pragma unroll
        for (int e = 0; e < 8; ++e) { s0[e] = 0.0f; s1[e] = 0.0f; }
        s0 = wmma_bf16(qa0, ld_b16(krow0 + 0  + 16 * hi), s0);
        s0 = wmma_bf16(qa1, ld_b16(krow0 + 32 + 16 * hi), s0);
        s1 = wmma_bf16(qa0, ld_b16(krow1 + 0  + 16 * hi), s1);
        s1 = wmma_bf16(qa1, ld_b16(krow1 + 32 + 16 * hi), s1);

        // ---- scale + causal mask, online softmax ----
        const int key0 = kbase + ln;
        const int key1 = key0 + 16;
#pragma unroll
        for (int v = 0; v < 8; ++v) {
            const int qr = m0 + v + 8 * hi;
            s0[v] = (key0 > qr) ? -1e30f : s0[v] * 0.125f;
            s1[v] = (key1 > qr) ? -1e30f : s1[v] * 0.125f;
        }
#pragma unroll
        for (int v = 0; v < 8; ++v) {
            float cmax = fmaxf(s0[v], s1[v]);
            cmax = fmaxf(cmax, __shfl_xor(cmax, 1, 32));
            cmax = fmaxf(cmax, __shfl_xor(cmax, 2, 32));
            cmax = fmaxf(cmax, __shfl_xor(cmax, 4, 32));
            cmax = fmaxf(cmax, __shfl_xor(cmax, 8, 32));
            float nm = fmaxf(rmax[v], cmax);
            float al = expf(rmax[v] - nm);
            rmax[v] = nm;
            float p0 = expf(s0[v] - nm);
            float p1 = expf(s1[v] - nm);
            float ts = p0 + p1;
            ts += __shfl_xor(ts, 1, 32);
            ts += __shfl_xor(ts, 2, 32);
            ts += __shfl_xor(ts, 4, 32);
            ts += __shfl_xor(ts, 8, 32);
            rsum[v] = rsum[v] * al + ts;
#pragma unroll
            for (int f = 0; f < 8; ++f) acc[f][v] *= al;
            sP[g][(v + 8 * hi) * 32 + ln]      = tobf(p0);
            sP[g][(v + 8 * hi) * 32 + 16 + ln] = tobf(p1);
        }
        __syncthreads();

        // ---- O += P[16,32] * V[32,128] ----
        v16bf pa = ld_a_bf16(&sP[g][(size_t)ln * 32], 0, hi);
#pragma unroll
        for (int f = 0; f < 8; ++f) {
            const __bf16* vrow =
                Vbase + (size_t)(f * 16 + ln) * S_ + kbase + 16 * hi;
            acc[f] = wmma_bf16(pa, ld_b16(vrow), acc[f]);
        }
        __syncthreads();
    }

    // ---- normalize by row sums, park in LDS ----
#pragma unroll
    for (int v = 0; v < 8; ++v) {
        float inv = 1.0f / rsum[v];
#pragma unroll
        for (int f = 0; f < 8; ++f)
            sO[g][(v + 8 * hi) * HD_ + f * 16 + ln] = acc[f][v] * inv;
    }
    __syncthreads();

    // ---- combine groups, RMSNorm over HD, scale, store bf16 ----
    const float lam = *lam_p;
    const int t    = threadIdx.x;
    const int row  = t >> 2;          // 16 rows, 4 threads each
    const int part = t & 3;           // 32 dims per thread
    float comb[32];
    float ssq = 0.0f;
#pragma unroll
    for (int j = 0; j < 32; ++j) {
        int d = part * 32 + j;
        float cv = sO[0][row * HD_ + d] - lam * sO[1][row * HD_ + d];
        comb[j] = cv;
        ssq += cv * cv;
    }
    sRed[t] = ssq;
    __syncthreads();
    float tot = sRed[row * 4 + 0] + sRed[row * 4 + 1] +
                sRed[row * 4 + 2] + sRed[row * 4 + 3];
    float sc = rsqrtf(tot * (1.0f / HD_) + 1e-6f) * ONE_MINUS_LI_F;
    __bf16* orow = O + ((size_t)b * S_ + m0 + row) * HID_ + h * HD_;
#pragma unroll
    for (int j = 0; j < 32; ++j) {
        int d = part * 32 + j;
        orow[d] = tobf(comb[j] * sc * gnorm_w[d]);
    }
}

// ---------------------------------------------------------------------------
// Host-side orchestration.
// ---------------------------------------------------------------------------
extern "C" void kernel_launch(void* const* d_in, const int* in_sizes, int n_in,
                              void* d_out, int out_size, void* d_ws, size_t ws_size,
                              hipStream_t stream) {
    (void)in_sizes; (void)n_in; (void)out_size; (void)ws_size;
    const float* q    = (const float*)d_in[0];
    const float* k    = (const float*)d_in[1];
    const float* v    = (const float*)d_in[2];
    const float* Wq   = (const float*)d_in[3];
    const float* Wk   = (const float*)d_in[4];
    const float* Wv   = (const float*)d_in[5];
    const float* Wo   = (const float*)d_in[6];
    const float* lq1  = (const float*)d_in[7];
    const float* lk1  = (const float*)d_in[8];
    const float* lq2  = (const float*)d_in[9];
    const float* lk2  = (const float*)d_in[10];
    const float* gw   = (const float*)d_in[11];
    const float* cosE = (const float*)d_in[12];
    const float* sinE = (const float*)d_in[13];
    // d_in[14] = causal mask; causality computed analytically in-kernel.
    float* out = (float*)d_out;

    // workspace layout (bytes); all offsets 8 MiB aligned
    const size_t ACT = (size_t)MROWS * HID_ * sizeof(__bf16);   // 16 MiB
    const size_t WB  = (size_t)HID_ * HID_ * sizeof(__bf16);    //  8 MiB
    char* w = (char*)d_ws;
    __bf16* actb = (__bf16*)(w);                    // activations (reused)
    __bf16* wb   = (__bf16*)(w + ACT);              // weights (reused)
    __bf16* proj = (__bf16*)(w + ACT + WB);         // projection scratch
    __bf16* QhP  = (__bf16*)(w + 2 * ACT + WB);
    __bf16* KhP  = (__bf16*)(w + 3 * ACT + WB);
    __bf16* VtP  = (__bf16*)(w + 4 * ACT + WB);
    __bf16* Obuf = (__bf16*)(w + 5 * ACT + WB);
    float*  lamP = (float*) (w + 6 * ACT + WB);

    const int cvtActGrid = (int)((ACT / sizeof(__bf16)) / 8 / 256);  // 4096
    const int cvtWGrid   = (int)((WB  / sizeof(__bf16)) / 8 / 256);  // 2048
    const int gemmGrid   = (MROWS / 64) * (HID_ / 128);              // 1024
    const int ropeGrid   = (B_ * 2 * NH_ * S_ * 32) / 256;           // 16384
    const int vtGrid     = (B_ * NH_ * HD_ * S_) / 256;              // 32768
    const int attnGrid   = B_ * NH_ * (S_ / 16);                     // 4096

    // Q projection + RoPE
    cvt_kernel<<<cvtActGrid, 256, 0, stream>>>(q, actb);
    cvt_kernel<<<cvtWGrid, 256, 0, stream>>>(Wq, wb);
    gemm_bf16<false><<<gemmGrid, 128, 0, stream>>>(actb, wb, proj);
    rope_kernel<<<ropeGrid, 256, 0, stream>>>(proj, cosE, sinE, QhP);
    // K projection + RoPE
    cvt_kernel<<<cvtActGrid, 256, 0, stream>>>(k, actb);
    cvt_kernel<<<cvtWGrid, 256, 0, stream>>>(Wk, wb);
    gemm_bf16<false><<<gemmGrid, 128, 0, stream>>>(actb, wb, proj);
    rope_kernel<<<ropeGrid, 256, 0, stream>>>(proj, cosE, sinE, KhP);
    // V projection + transpose
    cvt_kernel<<<cvtActGrid, 256, 0, stream>>>(v, actb);
    cvt_kernel<<<cvtWGrid, 256, 0, stream>>>(Wv, wb);
    gemm_bf16<false><<<gemmGrid, 128, 0, stream>>>(actb, wb, proj);
    vtrans_kernel<<<vtGrid, 256, 0, stream>>>(proj, VtP);
    // lambda scalar
    lambda_kernel<<<1, 128, 0, stream>>>(lq1, lk1, lq2, lk2, lamP);
    // flash differential attention -> Obuf (bf16, [B,S,HID] order)
    diffattn_kernel<<<attnGrid, 64, 0, stream>>>(QhP, KhP, VtP, lamP, gw, Obuf);
    // output projection -> d_out (fp32)
    cvt_kernel<<<cvtWGrid, 256, 0, stream>>>(Wo, wb);
    gemm_bf16<true><<<gemmGrid, 128, 0, stream>>>(Obuf, wb, out);
}
